// Model_61993557950767
// MI455X (gfx1250) — matrix operand order repaired and checked by reading, not compiled, not run on hardware
//
#include <hip/hip_runtime.h>
#include <hip/hip_bf16.h>
#include <math.h>

#define Bn 8
#define Cn 32
#define Kn 512

typedef __attribute__((ext_vector_type(2)))  float    v2f;
typedef __attribute__((ext_vector_type(8)))  float    v8f;
typedef __attribute__((ext_vector_type(16))) _Float16 v16h;

// ---------------- Stage 1: s = w_a*amp + w_p*phase (f32 + f16 copies) ----------------
__global__ void __launch_bounds__(256)
k_fuse(const float* __restrict__ amp, const float* __restrict__ ph,
       const float* __restrict__ wA, const float* __restrict__ wP,
       float* __restrict__ s32, _Float16* __restrict__ s16) {
    int idx = blockIdx.x * blockDim.x + threadIdx.x;
    if (idx >= Bn * Cn * Kn) return;
    float v = wA[0] * amp[idx] + wP[0] * ph[idx];
    s32[idx] = v;
    s16[idx] = (_Float16)v;
}

// ---------------- Stage 2: q[b][k] = sum_c s^2 ----------------
__global__ void __launch_bounds__(256)
k_q(const float* __restrict__ s32, float* __restrict__ q) {
    int idx = blockIdx.x * blockDim.x + threadIdx.x;
    if (idx >= Bn * Kn) return;
    int b = idx / Kn, k = idx % Kn;
    const float* p = s32 + b * Cn * Kn + k;
    float acc = 0.f;
#pragma unroll
    for (int c = 0; c < Cn; ++c) { float v = p[c * Kn]; acc += v * v; }
    q[idx] = acc;
}

// ---------------- Stage 3: batched Gram via WMMA, emit exp_dist ----------------
// One wave per 16x16 tile of G_b = S_b^T S_b (S_b is 32x512).
// dist = d_j^2 * (q_i + q_j - 2*G_ij); exp_dist = offdiag / (dist + 1e-10)
__global__ void __launch_bounds__(256)
k_gram(const float* __restrict__ s32, const _Float16* __restrict__ s16,
       const float* __restrict__ q, const float* __restrict__ A,
       float* __restrict__ exp_dist) {
    const int wave = threadIdx.x >> 5;
    const int lane = threadIdx.x & 31;
    const int tile = blockIdx.x * 8 + wave;   // 8192 tiles = 8 * 32 * 32
    const int b  = tile >> 10;
    const int ti = (tile >> 5) & 31;
    const int tj = tile & 31;
    const int i0 = ti * 16, j0 = tj * 16;
    const int m = lane & 15;                  // row within A-tile / col within B-tile
    const int h = lane >> 4;                  // lane half selects K pair / row offset

    v8f acc = {};

#if __has_builtin(__builtin_amdgcn_wmma_f32_16x16x4_f32)
    // Exact f32 path: 8 x V_WMMA_F32_16X16X4_F32 over the C=32 reduction.
    {
        const float* S = s32 + b * Cn * Kn;
#pragma unroll
        for (int c0 = 0; c0 < Cn; c0 += 4) {
            const int k0 = c0 + h * 2;        // lanes 0-15: K=0,1 ; lanes 16-31: K=2,3
            v2f a, bb;
            a[0]  = S[(k0 + 0) * Kn + i0 + m];
            a[1]  = S[(k0 + 1) * Kn + i0 + m];
            bb[0] = S[(k0 + 0) * Kn + j0 + m];
            bb[1] = S[(k0 + 1) * Kn + j0 + m];
            acc = __builtin_amdgcn_wmma_f32_16x16x4_f32(
                false, a, false, bb, (short)0, acc, false, false);
        }
    }
#else
    // Fallback: single f16 WMMA covers the whole K=32 reduction (codegen-confirmed).
    {
        const _Float16* S = s16 + b * Cn * Kn;
        v16h a, bb;
#pragma unroll
        for (int e = 0; e < 16; ++e) {
            const int v = e >> 1, pbit = e & 1;
            const int kA = (v < 4 ? 2 * v : 16 + 2 * (v - 4)) + h * 8 + pbit;
            a[e]  = S[kA * Kn + i0 + m];
            const int kB = (e < 8 ? e : 8 + e) + h * 8;
            bb[e] = S[kB * Kn + j0 + m];
        }
        acc = __builtin_amdgcn_wmma_f32_16x16x32_f16(
            false, a, false, bb, (short)0, acc, false, false);
    }
#endif

    const int j   = j0 + m;
    const float qj  = q[b * Kn + j];
    const float dj  = A[j * Kn + j];
    const float dsq = dj * dj;
#pragma unroll
    for (int v = 0; v < 8; ++v) {
        const int i = i0 + v + 8 * h;         // C/D layout: row = vgpr + 8*(lane/16)
        const float qi   = q[b * Kn + i];
        const float dist = dsq * (qi + qj - 2.0f * acc[v]);
        const float e    = (i == j) ? 0.0f : 1.0f / (dist + 1e-10f);
        exp_dist[(b * Kn + i) * Kn + j] = e;
    }
}

// ---------------- Stage 4: row max over j, per (b,i) ----------------
__global__ void __launch_bounds__(256)
k_rowmax(const float* __restrict__ exp_dist, float* __restrict__ rowmax) {
    const int wave = threadIdx.x >> 5;
    const int lane = threadIdx.x & 31;
    const int row  = blockIdx.x * 8 + wave;   // 4096 rows = B*K
    const float* p = exp_dist + (size_t)row * Kn;
    float mx = -INFINITY;
    for (int j = lane; j < Kn; j += 32) mx = fmaxf(mx, p[j]);
#pragma unroll
    for (int off = 16; off > 0; off >>= 1)
        mx = fmaxf(mx, __shfl_xor(mx, off, 32));
    if (lane == 0) rowmax[row] = mx;
}

// ---------------- Stage 5: Gumbel-argmax hard sample, mean over b ----------------
__device__ __forceinline__ float u01(unsigned x) {
    x ^= x >> 16; x *= 0x7feb352dU;
    x ^= x >> 15; x *= 0x846ca68bU;
    x ^= x >> 16;
    return ((float)(x >> 8) + 0.5f) * (1.0f / 16777216.0f);  // (0,1)
}

__global__ void __launch_bounds__(256)
k_sample(const float* __restrict__ exp_dist, const float* __restrict__ rowmax,
         float* __restrict__ out) {
    int idx = blockIdx.x * blockDim.x + threadIdx.x;  // K*K
    if (idx >= Kn * Kn) return;
    const int i = idx / Kn, j = idx % Kn;
    float acc = 0.f;
#pragma unroll
    for (int b = 0; b < Bn; ++b) {
        float p;
        if (i == j) {
            p = 0.99f;
        } else {
            const float e  = exp_dist[(b * Kn + i) * Kn + j];
            const float mx = rowmax[b * Kn + i];
            p = (e / mx) * 0.99f;
        }
        const float logit = __logf(p / (1.0f - p));
        const unsigned base = ((unsigned)((b * Kn + i) * Kn + j)) * 2u;
        const float g0 = -__logf(-__logf(u01(base ^ 0x9e3779b9u)));
        const float g1 = -__logf(-__logf(u01((base + 1u) ^ 0x85ebca6bu)));
        // hard sample of class 0: argmax(logit+g0, -logit+g1) == 0
        acc += ((logit + g0) >= (-logit + g1)) ? 1.0f : 0.0f;
    }
    out[idx] = acc * (1.0f / (float)Bn);
}

// ---------------- Launcher ----------------
extern "C" void kernel_launch(void* const* d_in, const int* in_sizes, int n_in,
                              void* d_out, int out_size, void* d_ws, size_t ws_size,
                              hipStream_t stream) {
    const float* amp = (const float*)d_in[0];
    const float* ph  = (const float*)d_in[1];
    const float* A   = (const float*)d_in[2];
    const float* wA  = (const float*)d_in[3];
    const float* wP  = (const float*)d_in[4];
    float* out = (float*)d_out;

    char* ws = (char*)d_ws;
    float*    s32 = (float*)(ws);                 //   524288 B : s (f32)
    _Float16* s16 = (_Float16*)(ws + 524288);     //   262144 B : s (f16)
    float*    q   = (float*)(ws + 786432);        //    16384 B : q[b][k]
    float*    rmx = (float*)(ws + 802816);        //    16384 B : rowmax[b][i]
    float*    ed  = (float*)(ws + 819200);        //  8388608 B : exp_dist[b][i][j]
    (void)in_sizes; (void)n_in; (void)out_size; (void)ws_size;

    k_fuse  <<<(Bn * Cn * Kn + 255) / 256, 256, 0, stream>>>(amp, ph, wA, wP, s32, s16);
    k_q     <<<(Bn * Kn + 255) / 256,      256, 0, stream>>>(s32, q);
    k_gram  <<<1024,                        256, 0, stream>>>(s32, s16, q, A, ed);
    k_rowmax<<<512,                         256, 0, stream>>>(ed, rmx);
    k_sample<<<(Kn * Kn + 255) / 256,       256, 0, stream>>>(ed, rmx, out);
}
